// PointNet2MSGCls_41609643163996
// MI455X (gfx1250) — compile-verified, matrix operand order checked by reading, not run on hardware
//
#include <hip/hip_runtime.h>

// ---------------------------------------------------------------------------
// PointNet++ MSG classification forward for gfx1250 (MI455X, wave32, WMMA).
// All MLP matmuls run on v_wmma_f32_16x16x32_bf16 (bf16 in, f32 accumulate).
// Weights are staged once into d_ws as zero-padded bf16 [cout][cinp] tiles.
// ---------------------------------------------------------------------------

typedef __attribute__((ext_vector_type(16))) __bf16 v16bf;
typedef __attribute__((ext_vector_type(8)))  float  v8f;

union FragU { v16bf v; unsigned u[8]; };

// Load a 16x32 bf16 WMMA operand fragment from a row-major [rows][ld] array.
// Per ISA 7.12.2 (16-bit A 16x32): lane&15 selects the row (A) / column (B),
// lane[4] selects K-halves {0..7,16..23} vs {8..15,24..31}; halves are packed
// as consecutive-K pairs, so each lane does 8 aligned u32 loads.
__device__ __forceinline__ v16bf ld_frag(const __bf16* __restrict__ p, int row,
                                         int k0, int ld) {
  const int lane = threadIdx.x & 31;
  const unsigned* p32 = reinterpret_cast<const unsigned*>(
      p + (size_t)row * ld + k0 + ((lane >> 4) << 3));
  FragU f;
#pragma unroll
  for (int j = 0; j < 4; ++j) f.u[j] = p32[j];        // K = kb+0..7
#pragma unroll
  for (int j = 4; j < 8; ++j) f.u[j] = p32[j + 4];    // K = kb+16..23
  return f.v;
}

// One SharedMLP layer on a 16-row LDS tile: sOut = relu(sIn * W^T + b).
// 4 waves split the output-column tiles; D layout: VGPR i -> row i + 8*lane[4].
template <int CIN, int COUT, bool RELU>
__device__ __forceinline__ void mma_layer(const __bf16* __restrict__ sIn,
                                          const __bf16* __restrict__ W,
                                          const float* __restrict__ bias,
                                          __bf16* __restrict__ sOut) {
  const int wave = threadIdx.x >> 5;
  const int lane = threadIdx.x & 31;
  const int r16 = lane & 15;
  for (int ct = wave; ct < COUT / 16; ct += 4) {
    v8f acc = {};
#pragma unroll
    for (int kc = 0; kc < CIN / 32; ++kc) {
      v16bf a = ld_frag(sIn, r16, kc * 32, CIN);
      v16bf b = ld_frag(W, ct * 16 + r16, kc * 32, CIN);
      acc = __builtin_amdgcn_wmma_f32_16x16x32_bf16(false, a, false, b,
                                                    (short)0, acc, false, false);
    }
    const int col = ct * 16 + r16;
    const float bs = bias[col];
    const int rbase = (lane < 16) ? 0 : 8;
#pragma unroll
    for (int i = 0; i < 8; ++i) {
      float v = acc[i] + bs;
      if (RELU) v = v > 0.f ? v : 0.f;
      sOut[(rbase + i) * COUT + col] = (__bf16)v;
    }
  }
}

// Final layer fused with max-pool over the 16 rows. pooled init 0 gives the
// relu-of-max == max-of-relu identity. Each column tile is owned by exactly
// one wave, so the pooled read-modify-write needs no atomics.
template <int CIN, int COUT>
__device__ __forceinline__ void mma_pool(const __bf16* __restrict__ sIn,
                                         const __bf16* __restrict__ W,
                                         const float* __restrict__ bias,
                                         float* __restrict__ sPool) {
  const int wave = threadIdx.x >> 5;
  const int lane = threadIdx.x & 31;
  const int r16 = lane & 15;
  for (int ct = wave; ct < COUT / 16; ct += 4) {
    v8f acc = {};
#pragma unroll
    for (int kc = 0; kc < CIN / 32; ++kc) {
      v16bf a = ld_frag(sIn, r16, kc * 32, CIN);
      v16bf b = ld_frag(W, ct * 16 + r16, kc * 32, CIN);
      acc = __builtin_amdgcn_wmma_f32_16x16x32_bf16(false, a, false, b,
                                                    (short)0, acc, false, false);
    }
    const int col = ct * 16 + r16;
    const float bs = bias[col];
    float m = acc[0] + bs;
#pragma unroll
    for (int i = 1; i < 8; ++i) m = fmaxf(m, acc[i] + bs);
    m = fmaxf(m, __shfl_xor(m, 16, 32));  // lanes l and l+16 share a column
    if (lane < 16) sPool[col] = fmaxf(sPool[col], m);
  }
}

// ---------------------------------------------------------------------------
// Stage f32 weights [cout][cin] -> zero-padded bf16 [coutp][cinp].
__global__ void convert_w_kernel(const float* __restrict__ src,
                                 __bf16* __restrict__ dst, int coutp, int cout,
                                 int cin, int cinp) {
  int i = blockIdx.x * blockDim.x + threadIdx.x;
  if (i >= coutp * cinp) return;
  int r = i / cinp, k = i % cinp;
  float v = (r < cout && k < cin) ? src[r * cin + k] : 0.f;
  dst[i] = (__bf16)v;
}

// points [B,3,N] f32 -> xyz [B,N,3] f32 for coalesced per-point gathers.
__global__ void transpose_points_kernel(const float* __restrict__ pts,
                                        float* __restrict__ xyz, int Bn,
                                        int Nn) {
  int i = blockIdx.x * blockDim.x + threadIdx.x;
  if (i >= Bn * Nn) return;
  int b = i / Nn, n = i % Nn;
#pragma unroll
  for (int c = 0; c < 3; ++c)
    xyz[(size_t)i * 3 + c] = pts[((size_t)b * 3 + c) * Nn + n];
}

// ---------------------------------------------------------------------------
// Farthest point sampling: one block per batch, min-dist array lives in LDS,
// first-index tie-break argmax to match jnp.argmax. Writes centroid xyz.
template <int N, int M, int THREADS>
__global__ __launch_bounds__(THREADS) void fps_kernel(
    const float* __restrict__ xyz, float* __restrict__ new_xyz) {
  const int b = blockIdx.x, tid = threadIdx.x;
  __shared__ float mind[N];
  __shared__ float rv[THREADS];
  __shared__ int ri[THREADS];
  __shared__ float cur[3];
  __shared__ int s_cur;
  if (tid == 0) s_cur = 0;
  for (int i = tid; i < N; i += THREADS) mind[i] = 1e10f;
  for (int j = 0; j < M; ++j) {
    __syncthreads();
    const int ci = s_cur;
    if (tid < 3) {
      float v = xyz[((size_t)b * N + ci) * 3 + tid];
      cur[tid] = v;
      new_xyz[((size_t)b * M + j) * 3 + tid] = v;
    }
    __syncthreads();
    const float cx = cur[0], cy = cur[1], cz = cur[2];
    float bd = -1.f;
    int bi = 0;
    for (int i = tid; i < N; i += THREADS) {
      const float dx = xyz[((size_t)b * N + i) * 3 + 0] - cx;
      const float dy = xyz[((size_t)b * N + i) * 3 + 1] - cy;
      const float dz = xyz[((size_t)b * N + i) * 3 + 2] - cz;
      float mm = fminf(mind[i], dx * dx + dy * dy + dz * dz);
      mind[i] = mm;
      if (mm > bd) { bd = mm; bi = i; }
    }
    rv[tid] = bd;
    ri[tid] = bi;
    __syncthreads();
    for (int s = THREADS / 2; s > 0; s >>= 1) {
      if (tid < s) {
        if (rv[tid + s] > rv[tid] ||
            (rv[tid + s] == rv[tid] && ri[tid + s] < ri[tid])) {
          rv[tid] = rv[tid + s];
          ri[tid] = ri[tid + s];
        }
      }
      __syncthreads();
    }
    if (tid == 0) s_cur = ri[0];
  }
}

// ---------------------------------------------------------------------------
// Fused SA-MSG scale: ball-query + gather + 3-layer WMMA MLP + max-pool.
// One block (4 waves) per (centroid, batch). Neighbor rows stream through LDS
// in 16-row tiles so everything stays on-chip.
template <int NSRC, int K, int CINF, int CINP, int C0, int C1, int C2>
__global__ __launch_bounds__(128) void sa_scale_kernel(
    const float* __restrict__ src_xyz, const __bf16* __restrict__ src_feat,
    const float* __restrict__ cent_xyz, int M, const __bf16* __restrict__ W0,
    const float* __restrict__ B0, const __bf16* __restrict__ W1,
    const float* __restrict__ B1, const __bf16* __restrict__ W2,
    const float* __restrict__ B2, __bf16* __restrict__ out_feat, int CTOT,
    int col_off, float r2) {
  const int b = blockIdx.y, c = blockIdx.x, tid = threadIdx.x;

  __shared__ int s_nbr[K];
  __shared__ __align__(16) __bf16 sX[16 * CINP];
  __shared__ __align__(16) __bf16 sH0[16 * C0];
  __shared__ __align__(16) __bf16 sH1[16 * C1];
  __shared__ float sPool[C2];

  const float cx = cent_xyz[((size_t)b * M + c) * 3 + 0];
  const float cy = cent_xyz[((size_t)b * M + c) * 3 + 1];
  const float cz = cent_xyz[((size_t)b * M + c) * 3 + 2];

  // Ball query on wave 0: first-K in index order via ballot scan; pad with
  // the first hit (centroid itself is always in-radius -> >=1 hit).
  if (tid < 32) {
    const int lane = tid;
    int cnt = 0;
    for (int base = 0; base < NSRC && cnt < K; base += 32) {
      const int i = base + lane;
      const float dx = src_xyz[((size_t)b * NSRC + i) * 3 + 0] - cx;
      const float dy = src_xyz[((size_t)b * NSRC + i) * 3 + 1] - cy;
      const float dz = src_xyz[((size_t)b * NSRC + i) * 3 + 2] - cz;
      unsigned mask =
          __builtin_amdgcn_ballot_w32(dx * dx + dy * dy + dz * dz < r2);
      while (mask && cnt < K) {
        const int bit = __ffs(mask) - 1;
        if (lane == 0) s_nbr[cnt] = base + bit;
        mask &= mask - 1;
        ++cnt;
      }
    }
    if (lane == 0) {
      const int first = (cnt > 0) ? s_nbr[0] : 0;
      for (int j = cnt; j < K; ++j) s_nbr[j] = first;
    }
  }
  for (int i = tid; i < C2; i += 128) sPool[i] = 0.f;
  __syncthreads();

  for (int rt = 0; rt < K / 16; ++rt) {
    // Gather 16 neighbor rows: [rel_xyz | feature | zero-pad] as bf16.
    for (int e = tid; e < 16 * CINP; e += 128) {
      const int r = e / CINP, k = e % CINP;
      const int s = s_nbr[rt * 16 + r];
      float v = 0.f;
      if (k < 3)
        v = src_xyz[((size_t)b * NSRC + s) * 3 + k] -
            (k == 0 ? cx : (k == 1 ? cy : cz));
      else if (k < 3 + CINF)
        v = (float)src_feat[(size_t)(b * NSRC + s) * CINF + (k - 3)];
      sX[e] = (__bf16)v;
    }
    __syncthreads();
    mma_layer<CINP, C0, true>(sX, W0, B0, sH0);
    __syncthreads();
    mma_layer<C0, C1, true>(sH0, W1, B1, sH1);
    __syncthreads();
    mma_pool<C1, C2>(sH1, W2, B2, sPool);
    __syncthreads();
  }
  for (int i = tid; i < C2; i += 128)
    out_feat[(size_t)(b * M + c) * CTOT + col_off + i] = (__bf16)sPool[i];
}

// ---------------------------------------------------------------------------
// Local MLP over the 128 SA2 centroids + global max-pool: one block per batch.
__global__ __launch_bounds__(128) void local_mlp_kernel(
    const float* __restrict__ xyz, const __bf16* __restrict__ feat,
    const __bf16* __restrict__ W0, const float* __restrict__ B0,
    const __bf16* __restrict__ W1, const float* __restrict__ B1,
    const __bf16* __restrict__ W2, const float* __restrict__ B2,
    __bf16* __restrict__ out) {
  const int b = blockIdx.x, tid = threadIdx.x;
  constexpr int MPTS = 128, CINF = 640, CINP = 672, C0 = 256, C1 = 512,
                C2 = 1024;
  __shared__ __align__(16) __bf16 sX[16 * CINP];
  __shared__ __align__(16) __bf16 sH0[16 * C0];
  __shared__ __align__(16) __bf16 sH1[16 * C1];
  __shared__ float sPool[C2];
  for (int i = tid; i < C2; i += 128) sPool[i] = 0.f;
  __syncthreads();
  for (int rt = 0; rt < MPTS / 16; ++rt) {
    for (int e = tid; e < 16 * CINP; e += 128) {
      const int r = e / CINP, k = e % CINP;
      const int p = rt * 16 + r;
      float v = 0.f;
      if (k < 3)
        v = xyz[((size_t)b * MPTS + p) * 3 + k];
      else if (k < 3 + CINF)
        v = (float)feat[(size_t)(b * MPTS + p) * CINF + (k - 3)];
      sX[e] = (__bf16)v;
    }
    __syncthreads();
    mma_layer<CINP, C0, true>(sX, W0, B0, sH0);
    __syncthreads();
    mma_layer<C0, C1, true>(sH0, W1, B1, sH1);
    __syncthreads();
    mma_pool<C1, C2>(sH1, W2, B2, sPool);
    __syncthreads();
  }
  for (int i = tid; i < C2; i += 128) out[(size_t)b * C2 + i] = (__bf16)sPool[i];
}

// ---------------------------------------------------------------------------
// Head: rows = 16 batches = exactly one WMMA M-tile. 1024->512->256->40.
__global__ __launch_bounds__(128) void head_kernel(
    const __bf16* __restrict__ xin, const __bf16* __restrict__ Wg0,
    const float* __restrict__ Bg0, const __bf16* __restrict__ Wg1,
    const float* __restrict__ Bg1, const __bf16* __restrict__ Wc,
    const float* __restrict__ Bc, float* __restrict__ out) {
  const int tid = threadIdx.x;
  __shared__ __align__(16) __bf16 sX[16 * 1024];
  __shared__ __align__(16) __bf16 sH0[16 * 512];
  __shared__ __align__(16) __bf16 sH1[16 * 256];
  for (int i = tid; i < 16 * 1024; i += 128) sX[i] = xin[i];
  __syncthreads();
  mma_layer<1024, 512, true>(sX, Wg0, Bg0, sH0);
  __syncthreads();
  mma_layer<512, 256, true>(sH0, Wg1, Bg1, sH1);
  __syncthreads();
  // cls layer: cout padded 40->48, no relu; drop cols >= 40.
  const int wave = tid >> 5, lane = tid & 31, r16 = lane & 15;
  for (int ct = wave; ct < 3; ct += 4) {
    v8f acc = {};
#pragma unroll
    for (int kc = 0; kc < 256 / 32; ++kc) {
      v16bf a = ld_frag(sH1, r16, kc * 32, 256);
      v16bf bm = ld_frag(Wc, ct * 16 + r16, kc * 32, 256);
      acc = __builtin_amdgcn_wmma_f32_16x16x32_bf16(false, a, false, bm,
                                                    (short)0, acc, false, false);
    }
    const int col = ct * 16 + r16;
    if (col < 40) {
      const float bs = Bc[col];
      const int rbase = (lane < 16) ? 0 : 8;
#pragma unroll
      for (int i = 0; i < 8; ++i) out[(rbase + i) * 40 + col] = acc[i] + bs;
    }
  }
}

// ---------------------------------------------------------------------------
extern "C" void kernel_launch(void* const* d_in, const int* in_sizes, int n_in,
                              void* d_out, int out_size, void* d_ws,
                              size_t ws_size, hipStream_t stream) {
  (void)in_sizes; (void)n_in; (void)out_size; (void)ws_size;
  constexpr int B = 16, N = 8192;
  char* ws = (char*)d_ws;
  size_t cur = 0;
  auto alloc = [&](size_t bytes) -> char* {
    cur = (cur + 255) & ~(size_t)255;
    char* p = ws + cur;
    cur += bytes;
    return p;
  };

  // Flattened input order: points; sa[0..1] x scale x layer {W,b}; local;
  // global; cls. Weight matrix i -> d_in[idx], bias -> d_in[idx+1].
  struct WDesc { int idx, coutp, cout, cin, cinp; };
  const WDesc wd[24] = {
      // SA1: cin 3->32 pad
      {1, 32, 32, 3, 32},    {3, 32, 32, 32, 32},   {5, 64, 64, 32, 32},
      {7, 64, 64, 3, 32},    {9, 64, 64, 64, 64},   {11, 128, 128, 64, 64},
      {13, 64, 64, 3, 32},   {15, 96, 96, 64, 64},  {17, 128, 128, 96, 96},
      // SA2: cin 323->352 pad
      {19, 64, 64, 323, 352},  {21, 64, 64, 64, 64},    {23, 128, 128, 64, 64},
      {25, 128, 128, 323, 352},{27, 128, 128, 128, 128},{29, 256, 256, 128, 128},
      {31, 128, 128, 323, 352},{33, 128, 128, 128, 128},{35, 256, 256, 128, 128},
      // local: cin 643->672 pad
      {37, 256, 256, 643, 672},{39, 512, 512, 256, 256},{41, 1024, 1024, 512, 512},
      // global + cls (cout 40->48 pad)
      {43, 512, 512, 1024, 1024},{45, 256, 256, 512, 512},{47, 48, 40, 256, 256}};

  __bf16* wp[24];
  for (int m = 0; m < 24; ++m) {
    const size_t elems = (size_t)wd[m].coutp * wd[m].cinp;
    wp[m] = (__bf16*)alloc(elems * 2);
    convert_w_kernel<<<((int)elems + 255) / 256, 256, 0, stream>>>(
        (const float*)d_in[wd[m].idx], wp[m], wd[m].coutp, wd[m].cout,
        wd[m].cin, wd[m].cinp);
  }

  float* xyz0 = (float*)alloc((size_t)B * N * 3 * 4);
  float* nxyz1 = (float*)alloc((size_t)B * 512 * 3 * 4);
  __bf16* feat1 = (__bf16*)alloc((size_t)B * 512 * 320 * 2);
  float* nxyz2 = (float*)alloc((size_t)B * 128 * 3 * 4);
  __bf16* feat2 = (__bf16*)alloc((size_t)B * 128 * 640 * 2);
  __bf16* lfeat = (__bf16*)alloc((size_t)B * 1024 * 2);

  transpose_points_kernel<<<(B * N + 255) / 256, 256, 0, stream>>>(
      (const float*)d_in[0], xyz0, B, N);

  fps_kernel<8192, 512, 1024><<<B, 1024, 0, stream>>>(xyz0, nxyz1);

#define BIAS(i) ((const float*)d_in[i])
  // SA level 1: radii {0.1,0.2,0.4}, k {16,32,128}, out 64|128|128 = 320
  sa_scale_kernel<8192, 16, 0, 32, 32, 32, 64><<<dim3(512, B), 128, 0, stream>>>(
      xyz0, (const __bf16*)nullptr, nxyz1, 512, wp[0], BIAS(2), wp[1], BIAS(4),
      wp[2], BIAS(6), feat1, 320, 0, 0.01f);
  sa_scale_kernel<8192, 32, 0, 32, 64, 64, 128><<<dim3(512, B), 128, 0, stream>>>(
      xyz0, (const __bf16*)nullptr, nxyz1, 512, wp[3], BIAS(8), wp[4], BIAS(10),
      wp[5], BIAS(12), feat1, 320, 64, 0.04f);
  sa_scale_kernel<8192, 128, 0, 32, 64, 96, 128><<<dim3(512, B), 128, 0, stream>>>(
      xyz0, (const __bf16*)nullptr, nxyz1, 512, wp[6], BIAS(14), wp[7], BIAS(16),
      wp[8], BIAS(18), feat1, 320, 192, 0.16f);

  fps_kernel<512, 128, 512><<<B, 512, 0, stream>>>(nxyz1, nxyz2);

  // SA level 2: radii {0.2,0.4,0.8}, k {32,64,128}, out 128|256|256 = 640
  sa_scale_kernel<512, 32, 320, 352, 64, 64, 128><<<dim3(128, B), 128, 0, stream>>>(
      nxyz1, feat1, nxyz2, 128, wp[9], BIAS(20), wp[10], BIAS(22), wp[11],
      BIAS(24), feat2, 640, 0, 0.04f);
  sa_scale_kernel<512, 64, 320, 352, 128, 128, 256><<<dim3(128, B), 128, 0, stream>>>(
      nxyz1, feat1, nxyz2, 128, wp[12], BIAS(26), wp[13], BIAS(28), wp[14],
      BIAS(30), feat2, 640, 128, 0.16f);
  sa_scale_kernel<512, 128, 320, 352, 128, 128, 256><<<dim3(128, B), 128, 0, stream>>>(
      nxyz1, feat1, nxyz2, 128, wp[15], BIAS(32), wp[16], BIAS(34), wp[17],
      BIAS(36), feat2, 640, 384, 0.64f);

  local_mlp_kernel<<<B, 128, 0, stream>>>(nxyz2, feat2, wp[18], BIAS(38),
                                          wp[19], BIAS(40), wp[20], BIAS(42),
                                          lfeat);

  head_kernel<<<1, 128, 0, stream>>>(lfeat, wp[21], BIAS(44), wp[22], BIAS(46),
                                     wp[23], BIAS(48), (float*)d_out);
#undef BIAS
}
